// FieldLstmEncoder_26259430047974
// MI455X (gfx1250) — compile-verified, matrix operand order checked by reading, not run on hardware
//
#include <hip/hip_runtime.h>
#include <hip/hip_bf16.h>

typedef __attribute__((ext_vector_type(16))) __bf16 v16bf;
typedef __attribute__((ext_vector_type(8)))  __bf16 v8bf;
typedef __attribute__((ext_vector_type(8)))  float  v8f;

#define BB   64
#define SS   512
#define UNID 1024
#define HD   1024
#define FLDD 256
#define NBLK 32          // persistent-kernel blocks (256 waves total)

__device__ __forceinline__ float sigmoidf_(float x) {
    return 1.0f / (1.0f + __expf(-x));
}

// WMMA bf16 16x16x32, f32 accumulate
__device__ __forceinline__ v8f wmma_bf16(v16bf a, v16bf b, v8f c) {
    return __builtin_amdgcn_wmma_f32_16x16x32_bf16(
        false, a, false, b, (short)0, c, false, false);
}

// Load one 16x32 bf16 operand tile (A row-major, or B from pre-transposed (N,K) weights).
// 16-bit operand layout: lane half 0 holds K {0..7,16..23}, half 1 holds {8..15,24..31}.
// Works for global or LDS base pointers (two 16B vector loads per lane).
__device__ __forceinline__ v16bf load_op16x32(const __bf16* __restrict__ base,
                                              int row, int ld, int kbase, int hf) {
    const __bf16* p = base + (size_t)row * ld + kbase + hf * 8;
    v8bf lo = *(const v8bf*)(p);
    v8bf hi = *(const v8bf*)(p + 16);
    v16bf r;
#pragma unroll
    for (int i = 0; i < 8; ++i) { r[i] = lo[i]; r[8 + i] = hi[i]; }
    return r;
}

// ---------------- conversion / init kernels ----------------

__global__ void k_cvt_bf16(const float* __restrict__ src, __bf16* __restrict__ dst, int n) {
    int i = blockIdx.x * blockDim.x + threadIdx.x;
    int stride = gridDim.x * blockDim.x;
    for (; i < n; i += stride) dst[i] = (__bf16)src[i];
}

// dst[n*K + k] = src[(row_off + k)*src_ld + n]   (transpose fp32 -> bf16, (K,N) -> (N,K))
__global__ void k_transpose_bf16(const float* __restrict__ src, __bf16* __restrict__ dst,
                                 int K, int N, int src_ld, int row_off) {
    int i = blockIdx.x * blockDim.x + threadIdx.x;
    int total = K * N;
    int stride = gridDim.x * blockDim.x;
    for (; i < total; i += stride) {
        int n = i / K, k = i - n * K;
        dst[(size_t)n * K + k] = (__bf16)src[(size_t)(row_off + k) * src_ld + n];
    }
}

__global__ void k_zero_f32(float* __restrict__ p, int n) {
    int i = blockIdx.x * blockDim.x + threadIdx.x;
    int stride = gridDim.x * blockDim.x;
    for (; i < n; i += stride) p[i] = 0.0f;
}

__global__ void k_zero_bf16(__bf16* __restrict__ p, int n) {
    int i = blockIdx.x * blockDim.x + threadIdx.x;
    int stride = gridDim.x * blockDim.x;
    for (; i < n; i += stride) p[i] = (__bf16)0.0f;
}

__global__ void k_zero_u32(unsigned* __restrict__ p, int n) {
    int i = blockIdx.x * blockDim.x + threadIdx.x;
    int stride = gridDim.x * blockDim.x;
    for (; i < n; i += stride) p[i] = 0u;
}

// ---------------- field projection: FT = sigmoid(r)*tanh(d), [r|d] = F @ W1 + bW1 ----------------
// Time-major M order: m = s*B + b; output stored in WMMA C-tile layout (tile, lane, 8).

__global__ void k_field_gemm(const __bf16* __restrict__ Fbf,   // (BS, FLD), (b,s) row order
                             const __bf16* __restrict__ W1T,   // (2H, FLD) transposed
                             const float*  __restrict__ bW1,   // (2H)
                             float*        __restrict__ FT_sw) // tiles (BS/16 * H/16) x 256
{
    int lane = threadIdx.x & 31;
    int gw   = (blockIdx.x * blockDim.x + threadIdx.x) >> 5;
    const int NT = HD / 16;                 // 64
    int mt = gw / NT;
    int nt = gw - mt * NT;
    int hf  = lane >> 4;
    int l15 = lane & 15;
    int m    = mt * 16 + l15;               // time-major row: m = s*B + b
    int arow = (m & (BB - 1)) * SS + (m >> 6);  // source row in (b,s) layout
    int ncol = nt * 16 + l15;

    v8f accr = {}; v8f accd = {};
#pragma unroll
    for (int kb = 0; kb < FLDD / 32; ++kb) {
        v16bf a  = load_op16x32(Fbf, arow,      FLDD, kb * 32, hf);
        v16bf br = load_op16x32(W1T, ncol,      FLDD, kb * 32, hf);
        v16bf bd = load_op16x32(W1T, HD + ncol, FLDD, kb * 32, hf);
        accr = wmma_bf16(a, br, accr);
        accd = wmma_bf16(a, bd, accd);
    }
    float biasr = bW1[ncol];
    float biasd = bW1[HD + ncol];
    v8f o;
#pragma unroll
    for (int v = 0; v < 8; ++v)
        o[v] = sigmoidf_(accr[v] + biasr) * tanhf(accd[v] + biasd);
    *(v8f*)(FT_sw + ((size_t)(mt * NT + nt) * 256) + lane * 8) = o;
}

// ---------------- input projection: Gx = X @ Wx + bW (bias folded), swizzled store ----------------

__global__ void k_gx_gemm(const __bf16* __restrict__ Xbf,   // (BS, UNI), (b,s) row order
                          const __bf16* __restrict__ WxT,   // (4H, UNI) transposed
                          const float*  __restrict__ bW,    // (4H)
                          __bf16*       __restrict__ Gx_sw) // tiles (BS/16 * 4H/16) x 256
{
    int lane = threadIdx.x & 31;
    int gw   = (blockIdx.x * blockDim.x + threadIdx.x) >> 5;
    const int NT = 4 * HD / 16;             // 256
    int mt = gw / NT;
    int nt = gw - mt * NT;
    int hf  = lane >> 4;
    int l15 = lane & 15;
    int m    = mt * 16 + l15;               // time-major row
    int arow = (m & (BB - 1)) * SS + (m >> 6);
    int ncol = nt * 16 + l15;

    v8f acc = {};
#pragma unroll 4
    for (int kb = 0; kb < UNID / 32; ++kb) {
        v16bf a = load_op16x32(Xbf, arow, UNID, kb * 32, hf);
        v16bf b = load_op16x32(WxT, ncol, UNID, kb * 32, hf);
        acc = wmma_bf16(a, b, acc);
    }
    float bias = bW[ncol];
    v8bf o;
#pragma unroll
    for (int v = 0; v < 8; ++v) o[v] = (__bf16)(acc[v] + bias);
    *(v8bf*)(Gx_sw + ((size_t)(mt * NT + nt) * 256) + lane * 8) = o;
}

// ---------------- persistent recurrence: all 512 steps in one kernel ----------------
// 32 blocks x 8 waves; wave (mt, nt) owns one 16x16 (batch x hidden) tile.
// All waves of a block share batch-tile mt -> stage the 16x1024 bf16 h-slab in LDS once/step.
// Device-wide sync between steps: per-step atomic arrival counters (agent scope).

__global__ void k_lstm_persistent(const __bf16* __restrict__ WhT,    // (4H, H) transposed
                                  const __bf16* __restrict__ Gx_sw,  // swizzled gates
                                  const float*  __restrict__ FT_sw,  // swizzled field term
                                  const int*    __restrict__ input_length,
                                  float*        __restrict__ h_sw,   // swizzled (B,H) f32
                                  float*        __restrict__ c_sw,   // swizzled (B,H) f32
                                  __bf16*       __restrict__ hbf0,   // row-major (B,H) bf16
                                  __bf16*       __restrict__ hbf1,
                                  float*        __restrict__ emit,   // (B,S,H) row-major
                                  unsigned*     __restrict__ sync_cnt) // [SS]
{
    __shared__ __bf16 Atile[16 * HD];       // 32 KB of 320 KB WGP LDS

    int lane = threadIdx.x & 31;
    int gw   = (blockIdx.x * blockDim.x + threadIdx.x) >> 5;
    const int NT = HD / 16;                 // 64
    int mt = gw / NT;                       // 0..3 (uniform within a block)
    int nt = gw - mt * NT;                  // 0..63
    int hf  = lane >> 4;
    int l15 = lane & 15;
    int ncol = nt * 16 + l15;               // hidden index

    // hoist per-row sequence lengths (constant over t)
    int lenv[8];
#pragma unroll
    for (int v = 0; v < 8; ++v) lenv[v] = input_length[mt * 16 + hf * 8 + v];

    size_t sidx = ((size_t)(mt * NT + nt) * 256) + lane * 8;   // state tile offset

    for (int t = 0; t < SS; ++t) {
        const __bf16* h_in  = (t & 1) ? hbf1 : hbf0;
        __bf16*       h_out = (t & 1) ? hbf0 : hbf1;

        // ---- stage A slab (rows mt*16 .. mt*16+15 of h_in) into LDS ----
        {
            const v8bf* src = (const v8bf*)(h_in + (size_t)(mt * 16) * HD);
            v8bf*       dst = (v8bf*)Atile;
            for (int i = threadIdx.x; i < 16 * HD / 8; i += blockDim.x) dst[i] = src[i];
        }
        __syncthreads();

        // ---- 4 WMMA chains (i,j,f,o) sharing the LDS A tile ----
        v8f ai = {}, aj = {}, af = {}, ao = {};
#pragma unroll 4
        for (int kb = 0; kb < HD / 32; ++kb) {
            v16bf a   = load_op16x32(Atile, l15, HD, kb * 32, hf);
            v16bf bi  = load_op16x32(WhT, 0 * HD + ncol, HD, kb * 32, hf);
            v16bf bj  = load_op16x32(WhT, 1 * HD + ncol, HD, kb * 32, hf);
            v16bf bf_ = load_op16x32(WhT, 2 * HD + ncol, HD, kb * 32, hf);
            v16bf bo  = load_op16x32(WhT, 3 * HD + ncol, HD, kb * 32, hf);
            ai = wmma_bf16(a, bi,  ai);
            aj = wmma_bf16(a, bj,  aj);
            af = wmma_bf16(a, bf_, af);
            ao = wmma_bf16(a, bo,  ao);
        }

        // ---- vectorized epilogue loads (all in WMMA C-tile layout) ----
        int mtile = t * 4 + mt;             // time-major M-tile of this step's rows
        v8bf gxi = *(const v8bf*)(Gx_sw + ((size_t)(mtile * 256 + 0 * NT + nt) * 256) + lane * 8);
        v8bf gxj = *(const v8bf*)(Gx_sw + ((size_t)(mtile * 256 + 1 * NT + nt) * 256) + lane * 8);
        v8bf gxf = *(const v8bf*)(Gx_sw + ((size_t)(mtile * 256 + 2 * NT + nt) * 256) + lane * 8);
        v8bf gxo = *(const v8bf*)(Gx_sw + ((size_t)(mtile * 256 + 3 * NT + nt) * 256) + lane * 8);
        v8f  ftv = *(const v8f*)(FT_sw + ((size_t)(mtile * NT + nt) * 256) + lane * 8);
        v8f  cold = *(const v8f*)(c_sw + sidx);
        v8f  hold = *(const v8f*)(h_sw + sidx);

        v8f cnew, hnew;
#pragma unroll
        for (int v = 0; v < 8; ++v) {
            float gi = ai[v] + (float)gxi[v];
            float gj = aj[v] + (float)gxj[v];
            float gf = af[v] + (float)gxf[v];
            float go = ao[v] + (float)gxo[v];
            float cn = sigmoidf_(gf + 1.0f) * cold[v] + sigmoidf_(gi) * tanhf(gj) + ftv[v];
            float hn = sigmoidf_(go) * tanhf(cn);
            bool  act = t < lenv[v];
            int   b   = mt * 16 + hf * 8 + v;
            emit[((size_t)b * SS + t) * HD + ncol] = act ? hn : 0.0f;
            hnew[v] = act ? hn : hold[v];
            cnew[v] = act ? cn : cold[v];
            h_out[(size_t)b * HD + ncol] = (__bf16)hnew[v];
        }
        *(v8f*)(c_sw + sidx) = cnew;
        *(v8f*)(h_sw + sidx) = hnew;

        // ---- device-wide step barrier (per-t arrival counter, agent scope) ----
        __threadfence();
        __syncthreads();
        if (threadIdx.x == 0) {
            __hip_atomic_fetch_add(&sync_cnt[t], 1u, __ATOMIC_ACQ_REL, __HIP_MEMORY_SCOPE_AGENT);
            while (__hip_atomic_load(&sync_cnt[t], __ATOMIC_ACQUIRE, __HIP_MEMORY_SCOPE_AGENT)
                   < (unsigned)gridDim.x) {
                __builtin_amdgcn_s_sleep(1);
            }
        }
        __syncthreads();
    }
}

// ---------------- final h/c de-swizzle into output tail ----------------

__global__ void k_final(const float* __restrict__ h_sw, const float* __restrict__ c_sw,
                        float* __restrict__ out_h, float* __restrict__ out_c) {
    int i = blockIdx.x * blockDim.x + threadIdx.x;
    if (i >= BB * HD) return;
    int b = i / HD, n = i - b * HD;
    int mt = b >> 4, bl = b & 15;
    int hf = bl >> 3, v = bl & 7;
    int nt = n >> 4, l15 = n & 15;
    int lane = l15 + (hf << 4);
    size_t idx = ((size_t)(mt * (HD / 16) + nt) * 256) + lane * 8 + v;
    out_h[i] = h_sw[idx];
    out_c[i] = c_sw[idx];
}

extern "C" void kernel_launch(void* const* d_in, const int* in_sizes, int n_in,
                              void* d_out, int out_size, void* d_ws, size_t ws_size,
                              hipStream_t stream)
{
    const float* X   = (const float*)d_in[0]; // (B,S,UNI)
    const float* F   = (const float*)d_in[1]; // (B,S,FLD)
    const int*   len = (const int*)d_in[2];   // (B)
    const float* W   = (const float*)d_in[3]; // (UNI+H, 4H)
    const float* bW  = (const float*)d_in[4]; // (4H)
    const float* W1  = (const float*)d_in[5]; // (FLD, 2H)
    const float* bW1 = (const float*)d_in[6]; // (2H)
    float* out = (float*)d_out;               // emit | h_fin | c_fin

    const size_t BS = (size_t)BB * SS;
    char* ws = (char*)d_ws;
    size_t off = 0;
    auto alloc = [&](size_t bytes) -> void* {
        void* p = ws + off;
        off += (bytes + 255) & ~(size_t)255;
        return p;
    };
    __bf16*   Xbf   = (__bf16*)alloc(BS * UNID * 2);
    __bf16*   Fbf   = (__bf16*)alloc(BS * FLDD * 2);
    __bf16*   WxT   = (__bf16*)alloc((size_t)4 * HD * UNID * 2);
    __bf16*   WhT   = (__bf16*)alloc((size_t)4 * HD * HD * 2);
    __bf16*   W1T   = (__bf16*)alloc((size_t)2 * HD * FLDD * 2);
    float*    FT_sw = (float*)alloc(BS * HD * 4);
    __bf16*   Gx_sw = (__bf16*)alloc(BS * (size_t)(4 * HD) * 2);
    float*    h_sw  = (float*)alloc((size_t)BB * HD * 4);
    float*    c_sw  = (float*)alloc((size_t)BB * HD * 4);
    __bf16*   hbf0  = (__bf16*)alloc((size_t)BB * HD * 2);
    __bf16*   hbf1  = (__bf16*)alloc((size_t)BB * HD * 2);
    unsigned* scnt  = (unsigned*)alloc(SS * sizeof(unsigned));

    // --- stage 0: convert / transpose / init ---
    k_cvt_bf16<<<2048, 256, 0, stream>>>(X, Xbf, (int)(BS * UNID));
    k_cvt_bf16<<<1024, 256, 0, stream>>>(F, Fbf, (int)(BS * FLDD));
    k_transpose_bf16<<<1024, 256, 0, stream>>>(W,  WxT, UNID, 4 * HD, 4 * HD, 0);
    k_transpose_bf16<<<1024, 256, 0, stream>>>(W,  WhT, HD,   4 * HD, 4 * HD, UNID);
    k_transpose_bf16<<<512,  256, 0, stream>>>(W1, W1T, FLDD, 2 * HD, 2 * HD, 0);
    k_zero_f32 <<<256, 256, 0, stream>>>(h_sw, BB * HD);
    k_zero_f32 <<<256, 256, 0, stream>>>(c_sw, BB * HD);
    k_zero_bf16<<<256, 256, 0, stream>>>(hbf0, BB * HD);
    k_zero_u32 <<<4,   256, 0, stream>>>(scnt, SS);

    // --- stage 1: field projection + gate activation (2 WMMA chains / wave) ---
    {
        size_t waves = (BS / 16) * (HD / 16);             // 2048 * 64
        k_field_gemm<<<(unsigned)(waves / 8), 256, 0, stream>>>(Fbf, W1T, bW1, FT_sw);
    }
    // --- stage 2: time-hoisted input projection Gx = X @ Wx + bW ---
    {
        size_t waves = (BS / 16) * ((size_t)4 * HD / 16); // 2048 * 256
        k_gx_gemm<<<(unsigned)(waves / 8), 256, 0, stream>>>(Xbf, WxT, bW, Gx_sw);
    }
    // --- stage 3: persistent recurrence, one kernel for all 512 steps ---
    k_lstm_persistent<<<NBLK, 256, 0, stream>>>(WhT, Gx_sw, FT_sw, len,
                                                h_sw, c_sw, hbf0, hbf1, out, scnt);
    // --- stage 4: final states (de-swizzle) ---
    k_final<<<(BB * HD + 255) / 256, 256, 0, stream>>>(h_sw, c_sw,
                                                       out + BS * HD,
                                                       out + BS * HD + (size_t)BB * HD);
}
